// SynthesisLayer_14998025797804
// MI455X (gfx1250) — matrix operand order repaired and checked by reading, not compiled, not run on hardware
//
#include <hip/hip_runtime.h>
#include <hip/hip_bf16.h>
#include <stdint.h>

#define B_SZ  8
#define CIN   512
#define COUT  512
#define WDIM  512
#define RES   64
#define NPIX  (RES*RES)

typedef __attribute__((ext_vector_type(16))) __bf16 bf16x16;
typedef __attribute__((ext_vector_type(8)))  __bf16 bf16x8;
typedef __attribute__((ext_vector_type(8)))  float  f32x8;

union Frag { bf16x16 full; bf16x8 half[2]; };

// Async global->LDS copy via inline CDNA5 asm (ISA 15.18.3, GLOBAL_LOAD_ASYNC_TO_LDS_B128).
// Tracked by ASYNCcnt; pair with "s_wait_asynccnt 0" before the workgroup barrier.
#define USE_ASYNC_LDS 1

// ---------------- kernel 1: styles[b][c] = dot(w[b], aff_w[c]) / sqrt(512) + aff_b[c]
__global__ void styles_kernel(const float* __restrict__ w,
                              const float* __restrict__ aff_w,
                              const float* __restrict__ aff_b,
                              float* __restrict__ styles) {
  int idx = blockIdx.x * 256 + threadIdx.x;      // 0..4095
  int b = idx >> 9, c = idx & 511;
  const float4* wr = (const float4*)(w + (size_t)b * WDIM);
  const float4* ar = (const float4*)(aff_w + (size_t)c * WDIM);
  float s = 0.f;
  #pragma unroll 4
  for (int i = 0; i < WDIM / 4; ++i) {
    float4 a = wr[i], bb = ar[i];
    s += a.x * bb.x + a.y * bb.y + a.z * bb.z + a.w * bb.w;
  }
  styles[idx] = s * 0.04419417382415922f + aff_b[c];  // 1/sqrt(512)
}

// ---------------- kernel 2: dcoefs[b][co] = rsqrt(sum_ci,t (weight*style)^2 + 1e-8), wave per output
__global__ void demod_kernel(const float* __restrict__ weight,
                             const float* __restrict__ styles,
                             float* __restrict__ dcoefs) {
  int wave = threadIdx.x >> 5, lane = threadIdx.x & 31;
  int gid = blockIdx.x * 8 + wave;               // 0..4095
  int b = gid >> 9, co = gid & 511;
  float sum = 0.f;
  for (int ci = lane; ci < CIN; ci += 32) {
    float s = styles[b * CIN + ci];
    const float* wr = weight + ((size_t)co * CIN + ci) * 9;
    #pragma unroll
    for (int t = 0; t < 9; ++t) { float v = wr[t] * s; sum += v * v; }
  }
  #pragma unroll
  for (int off = 16; off > 0; off >>= 1) sum += __shfl_down(sum, off, 32);
  if (lane == 0) dcoefs[gid] = rsqrtf(sum + 1e-8f);
}

// ---------------- kernel 3: weight -> bf16, permuted into WMMA A-fragment-major layout
// tile (t, mt=co/16, kc=ci/32) of 512 halves; lane = row + 16*((k>>3)&1); pos = ((k>>4)<<3)+(k&7)
__global__ void wcvt_kernel(const float* __restrict__ weight,
                            unsigned short* __restrict__ wbf) {
  int idx = blockIdx.x * 256 + threadIdx.x;      // 0..262143
  int co = idx >> 9, ci = idx & 511;
  int mt = co >> 4, r = co & 15, kc = ci >> 5, k = ci & 31;
  int grp = k >> 3;
  int lane = r + 16 * (grp & 1);
  int pos  = (grp >> 1) * 8 + (k & 7);
  const float* wr = weight + ((size_t)co * CIN + ci) * 9;
  #pragma unroll
  for (int t = 0; t < 9; ++t) {
    __bf16 h = (__bf16)wr[t];
    wbf[(((size_t)(t * 32 + mt) * 16) + kc) * 512 + lane * 16 + pos] =
        *(const unsigned short*)&h;
  }
}

// ---------------- kernel 4: modulate x, convert to bf16, and transpose to pixel-major chunks:
// xt[b][kc][pixel(4096)][cin_l(32)] so the conv kernel can stage with plain/async b128 copies.
#define XCELLB 80   // LDS cell stride in transpose kernel (bank stagger)
__global__ __launch_bounds__(256) void xmodT_kernel(const float* __restrict__ x,
                                                    const float* __restrict__ styles,
                                                    unsigned short* __restrict__ xt) {
  __shared__ __align__(16) unsigned char lt[64 * XCELLB];   // 5120 B
  const int tid = threadIdx.x;
  const int row = blockIdx.x & 63;
  const int kc  = (blockIdx.x >> 6) & 15;
  const int b   = blockIdx.x >> 10;

  const int cin_l = tid >> 3;            // 0..31
  const int colg  = (tid & 7) * 8;       // 0..56
  const int bc = b * CIN + kc * 32 + cin_l;
  const float s = styles[bc];
  const float4* px = (const float4*)(x + (size_t)bc * NPIX + row * RES + colg);
  float4 v0 = px[0], v1 = px[1];
  float vals[8] = {v0.x, v0.y, v0.z, v0.w, v1.x, v1.y, v1.z, v1.w};
  #pragma unroll
  for (int j = 0; j < 8; ++j) {
    __bf16 h = (__bf16)(vals[j] * s);
    *(unsigned short*)(lt + (colg + j) * XCELLB + cin_l * 2) = *(const unsigned short*)&h;
  }
  __syncthreads();
  const int col = tid >> 2, q = tid & 3;
  uint4 v = *(const uint4*)(lt + col * XCELLB + q * 16);
  *(uint4*)(xt + (((size_t)(b * 16 + kc) * NPIX) + row * RES + col) * 32 + q * 8) = v;
}

// ---------------- main kernel: implicit 3x3 conv via 9 shifted GEMMs, bf16 WMMA
// workgroup tile: 256 couts x 128 pixels (2 rows); 8 waves of 64x64 (mi=4, ni=4)
#define LDS_PIXW 68     // 64 cols + halo, padded
#define PIXB     80     // bytes per pixel cell: 64B data + 16B pad (bank stagger)
__global__ __launch_bounds__(256) void modconv_kernel(
    const unsigned short* __restrict__ xt,
    const unsigned short* __restrict__ wbf,
    const float* __restrict__ dcoefs,
    const float* __restrict__ bias,
    const float* __restrict__ noise,
    const float* __restrict__ nstr,
    float* __restrict__ out) {
  __shared__ __align__(16) unsigned char lds_x[4 * LDS_PIXW * PIXB];  // 21760 B

  const int tid  = threadIdx.x;
  const int lane = tid & 31;
  const int lc   = lane & 15;
  const int lh   = lane >> 4;
  const int wave = tid >> 5;
  const int wm   = wave >> 1;                    // 0..3 (M)
  const int wn   = wave & 1;                     // 0..1 (N row)

  const int bx   = blockIdx.x;                   // 0..511
  const int b    = bx >> 6;
  const int tile = bx & 63;
  const int ct   = tile >> 5;                    // cout tile (256 wide): 0..1
  const int pt   = tile & 31;                    // pixel tile (2 rows)
  const int r0   = pt * 2;
  const int row  = r0 + wn;
  const int mbase = ct * 256 + wm * 64;

  // zero LDS once (halo cols / out-of-image rows stay zero; valid cells rewritten each chunk)
  {
    uint4 z = {0, 0, 0, 0};
    for (int i = tid; i < (4 * LDS_PIXW * PIXB) / 16; i += 256) ((uint4*)lds_x)[i] = z;
  }
  __syncthreads();

  f32x8 acc[4][4];
  #pragma unroll
  for (int mi = 0; mi < 4; ++mi)
    #pragma unroll
    for (int ni = 0; ni < 4; ++ni) {
      f32x8 z = {0.f, 0.f, 0.f, 0.f, 0.f, 0.f, 0.f, 0.f};
      acc[mi][ni] = z;
    }

  for (int kc = 0; kc < CIN / 32; ++kc) {
    __syncthreads();
    // stage 4 rows x 64 cols x 32 cin (16 KB) with 16B copies from pixel-major xt
    const size_t xt_base = (size_t)(b * 16 + kc) * NPIX * 32;   // halves
    #pragma unroll
    for (int u = 0; u < 4; ++u) {
      int unit = tid + u * 256;                  // 0..1023 : (r, col, quarter)
      int r = unit >> 8;
      int rem = unit & 255;
      int col = rem >> 2;
      int q = rem & 3;
      int grow = r0 - 1 + r;
      if (grow >= 0 && grow < RES) {
        const unsigned short* src = xt + xt_base + (size_t)(grow * RES + col) * 32 + q * 8;
        unsigned char* dst = lds_x + (r * LDS_PIXW + 1 + col) * PIXB + q * 16;
#if USE_ASYNC_LDS
        unsigned int ldsoff = (unsigned int)(size_t)dst;   // flat->LDS: low 32 bits
        asm volatile("global_load_async_to_lds_b128 %0, %1, off"
                     :: "v"(ldsoff), "v"(src) : "memory");
#else
        *(uint4*)dst = *(const uint4*)src;
#endif
      }
    }
#if USE_ASYNC_LDS
    asm volatile("s_wait_asynccnt 0x0" ::: "memory");
#endif
    // prefetch next chunk's tile into caches while we compute this one
    if (kc + 1 < CIN / 32) {
      int r = tid >> 6;                          // 0..3
      int col = tid & 63;
      int grow = r0 - 1 + r;
      if (grow >= 0 && grow < RES)
        __builtin_prefetch(xt + xt_base + NPIX * 32 + (size_t)(grow * RES + col) * 32, 0, 1);
    }
    __syncthreads();

    #pragma unroll
    for (int kh = 0; kh < 3; ++kh) {
      #pragma unroll
      for (int kw = 0; kw < 3; ++kw) {
        const int t = kh * 3 + kw;
        Frag a[4];
        #pragma unroll
        for (int mi = 0; mi < 4; ++mi) {
          int mt = ct * 16 + wm * 4 + mi;
          const unsigned short* ap =
              wbf + (((size_t)(t * 32 + mt) * 16) + kc) * 512 + lane * 16;
          a[mi].half[0] = *(const bf16x8*)ap;         // 32B/lane, fully coalesced
          a[mi].half[1] = *(const bf16x8*)(ap + 8);
        }
        #pragma unroll
        for (int ni = 0; ni < 4; ++ni) {
          // B layout: lanes 0-15 hold K=0-7,16-23 (col = lc), lanes 16-31 hold K=8-15,24-31
          Frag bf;
          int pix = ni * 16 + lc + kw;                // halo offset folded (kw-1+1)
          const unsigned char* bp =
              lds_x + ((wn + kh) * LDS_PIXW + pix) * PIXB + lh * 16;
          bf.half[0] = *(const bf16x8*)bp;
          bf.half[1] = *(const bf16x8*)(bp + 32);
          #pragma unroll
          for (int mi = 0; mi < 4; ++mi) {
            acc[mi][ni] = __builtin_amdgcn_wmma_f32_16x16x32_bf16(
                false, a[mi].full, false, bf.full, (short)0, acc[mi][ni],
                false, false);
          }
        }
      }
    }
  }

  // epilogue: y = acc*dcoef + noise*ns + bias; lrelu * sqrt(2)
  const float ns = nstr[0];
  #pragma unroll
  for (int mi = 0; mi < 4; ++mi) {
    float dc[8], bs[8];
    #pragma unroll
    for (int v = 0; v < 8; ++v) {
      int m = mbase + mi * 16 + v + 8 * lh;          // f32 C/D layout: M = v + 8*lh
      dc[v] = dcoefs[b * COUT + m];
      bs[v] = bias[m];
    }
    #pragma unroll
    for (int ni = 0; ni < 4; ++ni) {
      int col = ni * 16 + lc;
      float nz = noise[b * NPIX + row * RES + col] * ns;
      #pragma unroll
      for (int v = 0; v < 8; ++v) {
        int m = mbase + mi * 16 + v + 8 * lh;
        float y = acc[mi][ni][v] * dc[v] + nz + bs[v];
        y = (y >= 0.f ? y : 0.2f * y) * 1.4142135623730951f;
        out[((size_t)(b * COUT + m)) * NPIX + row * RES + col] = y;
      }
    }
  }
}

extern "C" void kernel_launch(void* const* d_in, const int* in_sizes, int n_in,
                              void* d_out, int out_size, void* d_ws, size_t ws_size,
                              hipStream_t stream) {
  const float* x      = (const float*)d_in[0];
  const float* w      = (const float*)d_in[1];
  const float* weight = (const float*)d_in[2];
  const float* aff_w  = (const float*)d_in[3];
  const float* aff_b  = (const float*)d_in[4];
  const float* bias   = (const float*)d_in[5];
  const float* nstr   = (const float*)d_in[6];
  const float* noise  = (const float*)d_in[7];
  float* out = (float*)d_out;

  char* ws = (char*)d_ws;
  float* styles       = (float*)(ws);                       // 16 KB
  float* dcoefs       = (float*)(ws + 16384);               // 16 KB
  unsigned short* wbf = (unsigned short*)(ws + 32768);      // 9*512*512*2 = 4,718,592 B
  unsigned short* xt  = (unsigned short*)(ws + 32768 + (size_t)9 * COUT * CIN * 2);
  // xt: 8*16*4096*32*2 = 33,554,432 B ; total ws used ~38.3 MB

  styles_kernel<<<16,   256, 0, stream>>>(w, aff_w, aff_b, styles);
  demod_kernel <<<512,  256, 0, stream>>>(weight, styles, dcoefs);
  wcvt_kernel  <<<1024, 256, 0, stream>>>(weight, wbf);
  xmodT_kernel <<<8192, 256, 0, stream>>>(x, styles, xt);
  modconv_kernel<<<512, 256, 0, stream>>>(xt, wbf, dcoefs, bias, noise, nstr, out);
}
